// SwinBlockV1_1331439862005
// MI455X (gfx1250) — matrix operand order repaired and checked by reading, not compiled, not run on hardware
//
#include <hip/hip_runtime.h>
#include <hip/hip_bf16.h>
#include <math.h>

typedef __bf16 bf16;
typedef __attribute__((ext_vector_type(16))) __bf16 v16bf;
typedef __attribute__((ext_vector_type(8)))  float  v8f;

#define WSZ 7
#define NWIN 64          // windows per batch (8x8)
#define NTOK 49          // ws*ws
#define CDIM 192
#define NH 6
#define HD 32
#define BATCH 64
#define LSEQ 3136        // 56*56
#define KEEP 32
#define BK 2048          // BATCH*KEEP
#define MROWS 100352     // BK*NTOK (divisible by 16)
#define FF 768
#define QSZ 19267584     // BK*NH*NTOK*HD == MROWS*CDIM
#define SCALE_Q 0.17677669529663687f
#define EPS_LN 1e-5f

// ---- index of x[b, L, c] for window-flat index f, token n, channel c ----
__device__ __forceinline__ long x_index(int f, int n, int c) {
    int b  = f >> 6;
    int w  = f & 63;
    int wh = w >> 3, ww = w & 7;
    int i  = n / 7,  j  = n % 7;
    int Lp = (wh * 7 + i) * 56 + (ww * 7 + j);
    return ((long)b * LSEQ + Lp) * CDIM + c;
}

__device__ __forceinline__ v16bf zero16() {
    v16bf z;
#pragma unroll
    for (int e = 0; e < 16; ++e) z[e] = (bf16)0.0f;
    return z;
}

// A fragment: 16x32 bf16, row-major source. element e -> K=kBase+h*16+e.
// Two contiguous 16B loads.
__device__ __forceinline__ v16bf load_a16(const bf16* A, long rowOff, int kBase, int h, bool valid) {
    if (valid) {
        union { v16bf v; uint4 u[2]; } t;
        t.u[0] = *reinterpret_cast<const uint4*>(A + rowOff + kBase + h * 16);
        t.u[1] = *reinterpret_cast<const uint4*>(A + rowOff + kBase + h * 16 + 8);
        return t.v;
    }
    return zero16();
}

// B fragment: 32x16 bf16, B[k][n] = Wt[n][k] (Wt row-major (Nout,K)).
// lane column = lane&15, K = kBase + h*16 + e -> one contiguous 32B chunk.
__device__ __forceinline__ v16bf load_b16(const bf16* Wt, long colOff, int kBase, int h) {
    union { v16bf v; uint4 u[2]; } t;
    const uint4* p = reinterpret_cast<const uint4*>(Wt + colOff + kBase + h * 16);
    t.u[0] = p[0];
    t.u[1] = p[1];
    return t.v;
}

// ================= K0: weight conversion + rel-bias expansion + zero ent ====
__global__ void k_prep(const float* qkvw, const float* projw, const float* fc1w,
                       const float* fc2w, const float* reltab,
                       bf16* qkvw_b, bf16* projw_b, bf16* fc1w_b, bf16* fc2w_b,
                       float* relbias, float* out_ent) {
    int o = blockIdx.x * 256 + threadIdx.x;
    if (o < 110592) { qkvw_b[o] = (bf16)qkvw[o]; return; }
    o -= 110592;
    if (o < 36864)  { projw_b[o] = (bf16)projw[o]; return; }
    o -= 36864;
    if (o < 147456) { fc1w_b[o] = (bf16)fc1w[o]; return; }
    o -= 147456;
    if (o < 147456) { fc2w_b[o] = (bf16)fc2w[o]; return; }
    o -= 147456;
    if (o < NH * NTOK * NTOK) {
        int h = o / (NTOK * NTOK);
        int r = o % (NTOK * NTOK);
        int q = r / NTOK, k = r % NTOK;
        int qi = q / 7, qj = q % 7, ki = k / 7, kj = k % 7;
        int idx = (qi - ki + 6) * 13 + (qj - kj + 6);
        relbias[o] = reltab[idx * NH + h];
        return;
    }
    o -= NH * NTOK * NTOK;
    if (o < BATCH * NWIN) out_ent[o] = 0.0f;
}

// ===== K1: out = x for NON-kept windows only (kept windows written by FC2) ==
__global__ void __launch_bounds__(256) k_copy_win(const float4* __restrict__ x4, float4* __restrict__ out4,
                                                  const int* __restrict__ keepmask) {
    int f = blockIdx.x;
    if (keepmask[f]) return;
    int b  = f >> 6;
    int w  = f & 63;
    int wh = w >> 3, ww = w & 7;
    // window = 49 rows of 48 float4 (CDIM=192)
    for (int i = threadIdx.x; i < NTOK * 48; i += 256) {
        int n = i / 48, cp = i % 48;
        int ii = n / 7, jj = n % 7;
        long p = ((long)b * LSEQ + (long)((wh * 7 + ii) * 56 + (ww * 7 + jj))) * 48 + cp;
        out4[p] = x4[p];
    }
}

// ================= K2: local = softmax(mean over tokens) per window ========
__global__ void __launch_bounds__(256) k_local(const float* __restrict__ x, float* __restrict__ local) {
    __shared__ float red[256];
    int f = blockIdx.x;               // b*64 + win
    int c = threadIdx.x;
    float m = 0.0f;
    if (c < CDIM) {
        for (int n = 0; n < NTOK; ++n) m += x[x_index(f, n, c)];
        m *= (1.0f / 49.0f);
    }
    red[threadIdx.x] = (c < CDIM) ? m : -1e30f;
    __syncthreads();
    for (int s = 128; s > 0; s >>= 1) {
        if (threadIdx.x < s) red[threadIdx.x] = fmaxf(red[threadIdx.x], red[threadIdx.x + s]);
        __syncthreads();
    }
    float mx = red[0];
    __syncthreads();
    float e = (c < CDIM) ? __expf(m - mx) : 0.0f;
    red[threadIdx.x] = e;
    __syncthreads();
    for (int s = 128; s > 0; s >>= 1) {
        if (threadIdx.x < s) red[threadIdx.x] += red[threadIdx.x + s];
        __syncthreads();
    }
    if (c < CDIM) local[(long)f * CDIM + c] = e / red[0];
}

// ======= K3: KL + exact top-k (rank, tie-break lower idx) + keep mask ======
__global__ void __launch_bounds__(256) k_select(const float* __restrict__ local, int* __restrict__ keep_flat,
                                                int* __restrict__ keepmask) {
    __shared__ float glob[CDIM];
    __shared__ float klb[NWIN];
    __shared__ int   keepb[NWIN];
    int b = blockIdx.x, t = threadIdx.x;
    if (t < CDIM) {
        float s = 0.0f;
        for (int w = 0; w < NWIN; ++w) s += local[((long)b * NWIN + w) * CDIM + t];
        glob[t] = s * (1.0f / NWIN);
    }
    __syncthreads();
    if (t < NWIN) {
        float kl = 0.0f;
        const float* lp = local + ((long)b * NWIN + t) * CDIM;
        for (int c = 0; c < CDIM; ++c) {
            float l = lp[c];
            kl += l * __logf(l / (glob[c] + 1e-8f));
        }
        klb[t] = kl;
    }
    __syncthreads();
    if (t < NWIN) {
        float kl = klb[t];
        int rank = 0;
        for (int j = 0; j < NWIN; ++j) {
            float kj = klb[j];
            if (kj > kl || (kj == kl && j < t)) ++rank;
        }
        keepb[t] = (rank < KEEP) ? 1 : 0;
        keepmask[b * NWIN + t] = keepb[t];
    }
    __syncthreads();
    if (t < NWIN && keepb[t]) {
        int pos = 0;
        for (int j = 0; j < t; ++j) pos += keepb[j];
        keep_flat[b * KEEP + pos] = b * NWIN + t;
    }
}

// ================= K4: gather + LayerNorm1 -> bf16, wave per row ===========
__global__ void __launch_bounds__(256) k_ln1(const float* __restrict__ x, const int* __restrict__ keep_flat,
                                             const float* __restrict__ g, const float* __restrict__ bt,
                                             bf16* __restrict__ out) {
    int wave = threadIdx.x >> 5, lane = threadIdx.x & 31;
    int row = blockIdx.x * 8 + wave;
    if (row >= MROWS) return;
    int kwin = row / NTOK, n = row % NTOK;
    int f = keep_flat[kwin];
    float v[6];
    float s = 0.0f;
#pragma unroll
    for (int i = 0; i < 6; ++i) { v[i] = x[x_index(f, n, lane + 32 * i)]; s += v[i]; }
#pragma unroll
    for (int m = 16; m >= 1; m >>= 1) s += __shfl_xor(s, m, 32);
    float mu = s * (1.0f / CDIM);
    float var = 0.0f;
#pragma unroll
    for (int i = 0; i < 6; ++i) { float d = v[i] - mu; var += d * d; }
#pragma unroll
    for (int m = 16; m >= 1; m >>= 1) var += __shfl_xor(var, m, 32);
    float rs = rsqrtf(var * (1.0f / CDIM) + EPS_LN);
    bf16* op = out + (long)row * CDIM;
#pragma unroll
    for (int i = 0; i < 6; ++i) {
        int c = lane + 32 * i;
        op[c] = (bf16)((v[i] - mu) * rs * g[c] + bt[c]);
    }
}

// ================= generic WMMA GEMM: wave computes 16x64 strip ============
template <int KD, int ND, class Epi>
__global__ void __launch_bounds__(256) k_gemm(const bf16* __restrict__ A, const bf16* __restrict__ Wt, Epi epi) {
    constexpr int NSTRIPS = ND / 64;
    int wave = threadIdx.x >> 5, lane = threadIdx.x & 31;
    int strip = blockIdx.x * 8 + wave;
    if (strip >= (MROWS / 16) * NSTRIPS) return;
    int mTile = strip / NSTRIPS;
    int nBase = (strip % NSTRIPS) * 64;
    int h = lane >> 4, l15 = lane & 15;
    long rowOff = (long)(mTile * 16 + l15) * KD;

    v8f acc[4];
#pragma unroll
    for (int i = 0; i < 4; ++i)
#pragma unroll
        for (int e = 0; e < 8; ++e) acc[i][e] = 0.0f;

    for (int kb = 0; kb < KD; kb += 32) {
        if (kb + 32 < KD) __builtin_prefetch(A + rowOff + kb + 32, 0, 3);  // global_prefetch_b8
        v16bf a = load_a16(A, rowOff, kb, h, true);
#pragma unroll
        for (int nt = 0; nt < 4; ++nt) {
            v16bf b = load_b16(Wt, (long)(nBase + nt * 16 + l15) * KD, kb, h);
            acc[nt] = __builtin_amdgcn_wmma_f32_16x16x32_bf16(false, a, false, b,
                                                              (short)0, acc[nt], false, false);
        }
    }
#pragma unroll
    for (int nt = 0; nt < 4; ++nt)
#pragma unroll
        for (int r = 0; r < 8; ++r)
            epi(mTile * 16 + r + 8 * h, nBase + nt * 16 + l15, acc[nt][r]);
}

// ---- epilogues -------------------------------------------------------------
struct EpiQKV {
    const float* bias; bf16* q; bf16* k; bf16* v;
    __device__ void operator()(int row, int col, float acc) const {
        float val = acc + bias[col];
        int s = col / CDIM, rem = col % CDIM;
        int head = rem >> 5, d = rem & 31;
        int kwin = row / NTOK, n = row % NTOK;
        long off = (((long)kwin * NH + head) * NTOK + n) * HD + d;
        if (s == 0)      q[off] = (bf16)(val * SCALE_Q);
        else if (s == 1) k[off] = (bf16)val;
        else             v[off] = (bf16)val;
    }
};

struct EpiProj {
    const float* bias; const float* x; const int* keep_flat; float* xa;
    __device__ void operator()(int row, int col, float acc) const {
        int kwin = row / NTOK, n = row % NTOK;
        int f = keep_flat[kwin];
        xa[(long)row * CDIM + col] = acc + bias[col] + x[x_index(f, n, col)];
    }
};

struct EpiFC1 {
    const float* bias; bf16* hid;
    __device__ void operator()(int row, int col, float acc) const {
        float z = acc + bias[col];
        float gl = 0.5f * z * (1.0f + erff(z * 0.70710678118654752f));
        hid[(long)row * FF + col] = (bf16)gl;
    }
};

struct EpiFC2 {
    const float* bias; const float* xa; const bf16* n2; const int* keep_flat; float* out;
    __device__ void operator()(int row, int col, float acc) const {
        float val = acc + bias[col] + xa[(long)row * CDIM + col] + (float)n2[(long)row * CDIM + col];
        int kwin = row / NTOK, n = row % NTOK;
        int f = keep_flat[kwin];
        out[x_index(f, n, col)] = val;
    }
};

// ================= K6: attention, 1 wave per (window, head) ================
__global__ void __launch_bounds__(128) k_attn(const bf16* __restrict__ Q, const bf16* __restrict__ K,
                                              const bf16* __restrict__ V, const float* __restrict__ relbias,
                                              bf16* __restrict__ O) {
    __shared__ float S[4][64 * 64];
    int wave = threadIdx.x >> 5, lane = threadIdx.x & 31;
    int pair = blockIdx.x * 4 + wave;           // 0..12287
    int kwin = pair / NH, head = pair % NH;
    int h = lane >> 4, l15 = lane & 15;
    float* sm = S[wave];
    const bf16* qp = Q + ((long)kwin * NH + head) * NTOK * HD;
    const bf16* kp = K + ((long)kwin * NH + head) * NTOK * HD;
    const bf16* vp = V + ((long)kwin * NH + head) * NTOK * HD;
    const float* rb = relbias + (long)head * NTOK * NTOK;

    for (int i = lane; i < 64 * 64; i += 32) sm[i] = 0.0f;
    __syncthreads();

    // S = (q*scale) @ k^T + rel_bias   (q pre-scaled)
#pragma unroll
    for (int mt = 0; mt < 4; ++mt) {
        int rA = mt * 16 + l15;
        v16bf a = load_a16(qp, (long)rA * HD, 0, h, rA < NTOK);
#pragma unroll
        for (int nt = 0; nt < 4; ++nt) {
            int cB = nt * 16 + l15;
            v16bf b = (cB < NTOK) ? load_b16(kp, (long)cB * HD, 0, h) : zero16();
            v8f d;
#pragma unroll
            for (int e = 0; e < 8; ++e) d[e] = 0.0f;
            d = __builtin_amdgcn_wmma_f32_16x16x32_bf16(false, a, false, b, (short)0, d, false, false);
#pragma unroll
            for (int r = 0; r < 8; ++r) {
                int m = mt * 16 + r + 8 * h;
                int nn = nt * 16 + l15;
                if (m < NTOK && nn < NTOK) sm[m * 64 + nn] = d[r] + rb[m * NTOK + nn];
            }
        }
    }
    __syncthreads();

    // row softmax (cols 49..63 stay zero => free K padding for P@V)
    for (int m = lane; m < NTOK; m += 32) {
        float* rp = sm + m * 64;
        float mx = -1e30f;
        for (int nn = 0; nn < NTOK; ++nn) mx = fmaxf(mx, rp[nn]);
        float s = 0.0f;
        for (int nn = 0; nn < NTOK; ++nn) { float e = __expf(rp[nn] - mx); rp[nn] = e; s += e; }
        float inv = 1.0f / s;
        for (int nn = 0; nn < NTOK; ++nn) rp[nn] *= inv;
    }
    __syncthreads();

    // O = P @ V
#pragma unroll
    for (int mt = 0; mt < 4; ++mt) {
#pragma unroll
        for (int nt = 0; nt < 2; ++nt) {
            v8f acc;
#pragma unroll
            for (int e = 0; e < 8; ++e) acc[e] = 0.0f;
#pragma unroll
            for (int kt = 0; kt < 2; ++kt) {
                v16bf a;
                const float* rp = sm + (mt * 16 + l15) * 64;
#pragma unroll
                for (int e = 0; e < 16; ++e) {
                    int kk = kt * 32 + h * 16 + e;
                    a[e] = (bf16)rp[kk];
                }
                v16bf b;
                int col = nt * 16 + l15;
#pragma unroll
                for (int e = 0; e < 16; ++e) {
                    int kk = kt * 32 + h * 16 + e;
                    b[e] = (kk < NTOK) ? vp[kk * HD + col] : (bf16)0.0f;
                }
                acc = __builtin_amdgcn_wmma_f32_16x16x32_bf16(false, a, false, b, (short)0, acc, false, false);
            }
#pragma unroll
            for (int r = 0; r < 8; ++r) {
                int m = mt * 16 + r + 8 * h;
                if (m < NTOK)
                    O[((long)kwin * NTOK + m) * CDIM + head * HD + nt * 16 + l15] = (bf16)acc[r];
            }
        }
    }
}

// ================= K8: entropy + LayerNorm2 per kept window ================
__global__ void __launch_bounds__(256) k_ent_ln2(const float* __restrict__ xa, const int* __restrict__ keep_flat,
                                                 const float* __restrict__ g, const float* __restrict__ bt,
                                                 float* __restrict__ out_ent, bf16* __restrict__ n2) {
    __shared__ float cm[CDIM];
    __shared__ float red[256];
    int kwin = blockIdx.x, t = threadIdx.x;
    const float* base = xa + (long)kwin * NTOK * CDIM;
    if (t < CDIM) {
        float s = 0.0f;
        for (int n = 0; n < NTOK; ++n) s += base[n * CDIM + t];
        cm[t] = s * (1.0f / 49.0f);
    }
    __syncthreads();
    float m = (t < CDIM) ? cm[t] : -1e30f;
    red[t] = m; __syncthreads();
    for (int s = 128; s > 0; s >>= 1) { if (t < s) red[t] = fmaxf(red[t], red[t + s]); __syncthreads(); }
    float mx = red[0]; __syncthreads();
    float e = (t < CDIM) ? __expf(m - mx) : 0.0f;
    red[t] = e; __syncthreads();
    for (int s = 128; s > 0; s >>= 1) { if (t < s) red[t] += red[t + s]; __syncthreads(); }
    float sum = red[0]; __syncthreads();
    float p = e / sum;
    red[t] = (t < CDIM) ? (-p * __logf(p + 1e-8f)) : 0.0f;
    __syncthreads();
    for (int s = 128; s > 0; s >>= 1) { if (t < s) red[t] += red[t + s]; __syncthreads(); }
    if (t == 0) out_ent[keep_flat[kwin]] = red[0];

    // LN2: wave per row
    int wave = t >> 5, lane = t & 31;
    for (int n = wave; n < NTOK; n += 8) {
        const float* rp = base + n * CDIM;
        float v[6];
        float s2 = 0.0f;
#pragma unroll
        for (int i = 0; i < 6; ++i) { v[i] = rp[lane + 32 * i]; s2 += v[i]; }
#pragma unroll
        for (int mm = 16; mm >= 1; mm >>= 1) s2 += __shfl_xor(s2, mm, 32);
        float mu = s2 * (1.0f / CDIM);
        float var = 0.0f;
#pragma unroll
        for (int i = 0; i < 6; ++i) { float d = v[i] - mu; var += d * d; }
#pragma unroll
        for (int mm = 16; mm >= 1; mm >>= 1) var += __shfl_xor(var, mm, 32);
        float rs = rsqrtf(var * (1.0f / CDIM) + EPS_LN);
        bf16* op = n2 + ((long)kwin * NTOK + n) * CDIM;
#pragma unroll
        for (int i = 0; i < 6; ++i) {
            int c = lane + 32 * i;
            op[c] = (bf16)((v[i] - mu) * rs * g[c] + bt[c]);
        }
    }
}

// ============================================================================
extern "C" void kernel_launch(void* const* d_in, const int* in_sizes, int n_in,
                              void* d_out, int out_size, void* d_ws, size_t ws_size,
                              hipStream_t stream) {
    (void)in_sizes; (void)n_in; (void)out_size; (void)ws_size;
    const float* x      = (const float*)d_in[0];
    const float* n1g    = (const float*)d_in[3];
    const float* n1b    = (const float*)d_in[4];
    const float* n2g    = (const float*)d_in[5];
    const float* n2b    = (const float*)d_in[6];
    const float* qkvw   = (const float*)d_in[7];
    const float* qkvb   = (const float*)d_in[8];
    const float* projw  = (const float*)d_in[9];
    const float* projb  = (const float*)d_in[10];
    const float* reltab = (const float*)d_in[11];
    const float* fc1w   = (const float*)d_in[12];
    const float* fc1b   = (const float*)d_in[13];
    const float* fc2w   = (const float*)d_in[14];
    const float* fc2b   = (const float*)d_in[15];

    float* out     = (float*)d_out;
    float* out_ent = out + (size_t)BATCH * LSEQ * CDIM;

    char* ws = (char*)d_ws;
    size_t off = 0;
    auto alloc = [&](size_t bytes) -> void* {
        void* p = ws + off;
        off = (off + bytes + 255) & ~(size_t)255;
        return p;
    };
    float* localb  = (float*)alloc((size_t)BATCH * NWIN * CDIM * 4);
    int*   keepf   = (int*)  alloc((size_t)BK * 4);
    int*   keepm   = (int*)  alloc((size_t)BATCH * NWIN * 4);
    bf16*  qkvw_b  = (bf16*) alloc((size_t)110592 * 2);
    bf16*  projw_b = (bf16*) alloc((size_t)36864 * 2);
    bf16*  fc1w_b  = (bf16*) alloc((size_t)147456 * 2);
    bf16*  fc2w_b  = (bf16*) alloc((size_t)147456 * 2);
    float* relbias = (float*)alloc((size_t)NH * NTOK * NTOK * 4);
    bf16*  Abuf    = (bf16*) alloc((size_t)MROWS * CDIM * 2);   // LN1 out, later n2
    bf16*  BIGH    = (bf16*) alloc((size_t)MROWS * FF * 2);     // qkv, later mlp hidden
    bf16*  Obuf    = (bf16*) alloc((size_t)MROWS * CDIM * 2);   // attention out
    float* XA      = (float*)alloc((size_t)MROWS * CDIM * 4);   // x_attn (f32)
    bf16* Qb = BIGH, *Kb = BIGH + QSZ, *Vb = BIGH + 2 * (size_t)QSZ;

    // 0: weights->bf16, rel bias expand, zero entropy output
    k_prep<<<1801, 256, 0, stream>>>(qkvw, projw, fc1w, fc2w, reltab,
                                     qkvw_b, projw_b, fc1w_b, fc2w_b, relbias, out_ent);
    // 1: local softmax of window means
    k_local<<<BATCH * NWIN, 256, 0, stream>>>(x, localb);
    // 2: KL + top-k selection (+ per-window keep mask)
    k_select<<<BATCH, 256, 0, stream>>>(localb, keepf, keepm);
    // 3: out = x for non-kept windows only (kept windows written by FC2)
    k_copy_win<<<BATCH * NWIN, 256, 0, stream>>>((const float4*)x, (float4*)out, keepm);
    // 4: gather + LN1 -> bf16
    k_ln1<<<MROWS / 8, 256, 0, stream>>>(x, keepf, n1g, n1b, Abuf);
    // 5: QKV GEMM (M=100352, K=192, N=576)
    k_gemm<CDIM, 576, EpiQKV><<<(MROWS / 16) * 9 / 8, 256, 0, stream>>>(
        Abuf, qkvw_b, EpiQKV{qkvb, Qb, Kb, Vb});
    // 6: attention
    k_attn<<<BK * NH / 4, 128, 0, stream>>>(Qb, Kb, Vb, relbias, Obuf);
    // 7: proj GEMM + residual -> XA (f32)
    k_gemm<CDIM, CDIM, EpiProj><<<(MROWS / 16) * 3 / 8, 256, 0, stream>>>(
        Obuf, projw_b, EpiProj{projb, x, keepf, XA});
    // 8: entropy + LN2 -> Abuf (n2, bf16)
    k_ent_ln2<<<BK, 256, 0, stream>>>(XA, keepf, n2g, n2b, out_ent, Abuf);
    // 9: fc1 GEMM + GELU -> BIGH (bf16)
    k_gemm<CDIM, FF, EpiFC1><<<(MROWS / 16) * 12 / 8, 256, 0, stream>>>(
        Abuf, fc1w_b, EpiFC1{fc1b, BIGH});
    // 10: fc2 GEMM + x_attn + n2, scatter into d_out
    k_gemm<FF, CDIM, EpiFC2><<<(MROWS / 16) * 3 / 8, 256, 0, stream>>>(
        BIGH, fc2w_b, EpiFC2{fc2b, XA, Abuf, keepf, out});
}
